// Denoising_Model_24764781429262
// MI455X (gfx1250) — matrix operand order repaired and checked by reading, not compiled, not run on hardware
//
#include <hip/hip_runtime.h>
#include <hip/hip_bf16.h>
#include <cstdint>

// ---------------------------------------------------------------------------
// Types for CDNA5 WMMA (wave32)
// ---------------------------------------------------------------------------
typedef __attribute__((ext_vector_type(16))) __bf16 bf16x16;
typedef __attribute__((ext_vector_type(8)))  __bf16 bf16x8;
typedef __attribute__((ext_vector_type(8)))  float  f32x8;

#define HC   256
#define KPAD 288   // 264 padded up to multiple of 32 (also covers 136)

static inline int cdiv(int a, int b) { return (a + b - 1) / b; }

__device__ __forceinline__ float elu1(float v) {
    return v > 0.f ? v : (__expf(v) - 1.f);
}

// ---------------------------------------------------------------------------
// Time embedding: sinusoidal(128) -> Linear(128,128)+ELU -> Linear(128,256)
// One block, 128 threads. Writes temb[256].
// ---------------------------------------------------------------------------
__global__ __launch_bounds__(128) void time_embed_kernel(
    const float* __restrict__ t_in,
    const float* __restrict__ w1, const float* __restrict__ b1,
    const float* __restrict__ w2, const float* __restrict__ b2,
    float* __restrict__ temb)
{
    __shared__ float emb[128];
    __shared__ float hid[128];
    const int i = threadIdx.x;
    const float xs = t_in[0] * 4.0f;   // t/num_steps*num_steps*rescale == t*4
    if (i < 64) {
        // freqs = exp(i * (-ln(10000)/63))
        float f = __expf((float)i * (-9.210340371976184f / 63.0f));
        float ang = xs * f;
        emb[i]      = __sinf(ang);
        emb[i + 64] = __cosf(ang);
    }
    __syncthreads();
    float acc = b1[i];
    #pragma unroll 4
    for (int k = 0; k < 128; ++k) acc += emb[k] * w1[k * 128 + i];
    hid[i] = elu1(acc);
    __syncthreads();
    #pragma unroll
    for (int rep = 0; rep < 2; ++rep) {
        int o = i + rep * 128;
        float a2 = b2[o];
        #pragma unroll 4
        for (int k = 0; k < 128; ++k) a2 += hid[k] * w2[k * HC + o];
        temb[o] = a2;
    }
}

// ---------------------------------------------------------------------------
// Convert W [K, Ncols] f32 -> Wt [Ncols, Kpad] bf16, zero-padded along K.
// ---------------------------------------------------------------------------
__global__ void convert_wt_kernel(const float* __restrict__ W,
                                  __bf16* __restrict__ Wt,
                                  int K, int Ncols, int Kpad)
{
    int t = blockIdx.x * blockDim.x + threadIdx.x;
    if (t >= Ncols * Kpad) return;
    int c = t / Kpad;
    int k = t - c * Kpad;
    float v = (k < K) ? W[(size_t)k * Ncols + c] : 0.f;
    Wt[t] = (__bf16)v;
}

// ---------------------------------------------------------------------------
// Build initial A (bf16, [N, KPAD]): concat(x[128], q[8]) + zero pad.
// ---------------------------------------------------------------------------
__global__ void build_a0_kernel(const float* __restrict__ x,
                                const float* __restrict__ q,
                                __bf16* __restrict__ Ab, int N)
{
    int t = blockIdx.x * blockDim.x + threadIdx.x;
    if (t >= N * KPAD) return;
    int n = t / KPAD;
    int j = t - n * KPAD;
    float v = (j < 128) ? x[(size_t)n * 128 + j]
            : (j < 136) ? q[(size_t)n * 8 + (j - 128)]
            : 0.f;
    Ab[t] = (__bf16)v;
}

// ---------------------------------------------------------------------------
// WMMA GEMM: C[M, Ncols] (f32) = A[M, lda] (bf16) x Bt[Ncols, ldb]^T (bf16)
//
// Block = 256 threads = 8 waves, covering a 16 x 128 output slab.
// The 16 x KPAD A tile is staged once per block into LDS with CDNA5 async
// global->LDS copies (ASYNCcnt / s_wait_asynccnt), then shared by all 8
// waves; per-wave B fragments stream from global (weights live in WGP$/L2).
//
// Fragment layouts (16-bit, wave32):
//  A 16x32: lanes 0-15 hold K 0..7 & 16..23 (2/VGPR), lanes 16-31 K 8..15 &
//  24..31, row = lane%16.  B 32x16: same K split as A (not contiguous),
//  col = lane%16.  C/D: VGPR i -> row half*8+i, col = lane%16.
// ---------------------------------------------------------------------------
#define ACH_PER_ROW (KPAD / 8)            // 16B chunks per A row (36)
#define ACHUNKS     (16 * ACH_PER_ROW)    // 576 chunks = 9216 B

__global__ __launch_bounds__(256) void wmma_gemm_bf16_kernel(
    const __bf16* __restrict__ A, int lda,
    const __bf16* __restrict__ Bt, int ldb,
    float* __restrict__ C, int ldc,
    int M, int Ncols, int K)
{
    __shared__ __align__(128) __bf16 Atile[16 * KPAD];

    const int wave = threadIdx.x >> 5;
    const int lane = threadIdx.x & 31;
    const int half = lane >> 4;
    const int r    = lane & 15;
    const int tileM = blockIdx.x * 16;
    const int tileN = blockIdx.y * 128 + wave * 16;

    // ---- cooperative async copy of the 16 x KPAD A tile into LDS ----
    for (int c = threadIdx.x; c < ACHUNKS; c += blockDim.x) {
        int row = c / ACH_PER_ROW;
        int col = c - row * ACH_PER_ROW;          // 16-byte chunk index in row
        int rg  = tileM + row;
        if (rg >= M) rg = M - 1;                  // clamp (stores predicated)
        const __bf16* gsrc = A + (size_t)rg * lda + col * 8;
        unsigned ldst = (unsigned)(size_t)(&Atile[row * KPAD + col * 8]);
        asm volatile("global_load_async_to_lds_b128 %0, %1, off"
                     :: "v"(ldst), "v"(gsrc) : "memory");
    }
    asm volatile("s_wait_asynccnt 0" ::: "memory");
    __syncthreads();

    if (tileN >= Ncols) return;

    const __bf16* __restrict__ arow = &Atile[r * KPAD];
    const __bf16* __restrict__ bcol = Bt + (size_t)(tileN + r) * ldb;

    f32x8 acc = {};
    for (int kk = 0; kk < K; kk += 32) {
        bf16x8 alo = *(const bf16x8*)(arow + kk + half * 8);
        bf16x8 ahi = *(const bf16x8*)(arow + kk + half * 8 + 16);
        bf16x16 a = __builtin_shufflevector(alo, ahi,
            0,1,2,3,4,5,6,7,8,9,10,11,12,13,14,15);
        bf16x8 blo = *(const bf16x8*)(bcol + kk + half * 8);
        bf16x8 bhi = *(const bf16x8*)(bcol + kk + half * 8 + 16);
        bf16x16 b = __builtin_shufflevector(blo, bhi,
            0,1,2,3,4,5,6,7,8,9,10,11,12,13,14,15);
        acc = __builtin_amdgcn_wmma_f32_16x16x32_bf16(
            false, a, false, b, (short)0, acc, false, false);
    }

    float* __restrict__ crow = C + (size_t)(tileM + half * 8) * ldc + tileN + r;
    #pragma unroll
    for (int i = 0; i < 8; ++i) {
        if (tileM + half * 8 + i < M) crow[(size_t)i * ldc] = acc[i];
    }
}

// ---------------------------------------------------------------------------
// Per-node attention logits: asrc[n,h] = <hw[n,h,:], att_src[h,:]>
// ---------------------------------------------------------------------------
__global__ void node_alpha_kernel(const float* __restrict__ hw,
                                  const float* __restrict__ att_src,
                                  const float* __restrict__ att_dst,
                                  float* __restrict__ asrc,
                                  float* __restrict__ adst, int N)
{
    int t = blockIdx.x * blockDim.x + threadIdx.x;
    if (t >= N * 8) return;
    int n = t >> 3, h = t & 7;
    const float* row = hw + (size_t)n * HC + h * 32;
    float s = 0.f, d = 0.f;
    #pragma unroll 8
    for (int c = 0; c < 32; ++c) {
        float v = row[c];
        s += v * att_src[h * 32 + c];
        d += v * att_dst[h * 32 + c];
    }
    asrc[t] = s;
    adst[t] = d;
}

// ---------------------------------------------------------------------------
// Zero conv accumulator; init segmax=-inf, denom=0.
// ---------------------------------------------------------------------------
__global__ void init_accum_kernel(float* __restrict__ conv,
                                  float* __restrict__ segmax,
                                  float* __restrict__ denom, int N)
{
    int t = blockIdx.x * blockDim.x + threadIdx.x;
    if (t < N * HC) conv[t] = 0.f;
    if (t < N * 8) { segmax[t] = -3.0e38f; denom[t] = 0.f; }
}

// ---------------------------------------------------------------------------
// Edge pass 1: eh = leaky_relu(asrc[src]+adst[dst], 0.2); atomicMax segmax[dst]
// ---------------------------------------------------------------------------
__global__ void edge_pass1_kernel(const long long* __restrict__ adj, int E, int Etot,
                                  const float* __restrict__ asrc,
                                  const float* __restrict__ adst,
                                  float* __restrict__ eh,
                                  float* __restrict__ segmax)
{
    int t = blockIdx.x * blockDim.x + threadIdx.x;
    if (t >= Etot * 8) return;
    int e = t >> 3, h = t & 7;
    long long s, d;
    if (e < E) { s = adj[e]; d = adj[(size_t)E + e]; } else { s = d = e - E; }
    float v = asrc[s * 8 + h] + adst[d * 8 + h];
    v = v > 0.f ? v : 0.2f * v;
    eh[t] = v;
    atomicMax(&segmax[d * 8 + h], v);
}

// ---------------------------------------------------------------------------
// Edge pass 2: ex = exp(eh - segmax[dst]); atomicAdd denom[dst]. ex overwrites eh.
// ---------------------------------------------------------------------------
__global__ void edge_pass2_kernel(const long long* __restrict__ adj, int E, int Etot,
                                  float* __restrict__ eh,
                                  const float* __restrict__ segmax,
                                  float* __restrict__ denom)
{
    int t = blockIdx.x * blockDim.x + threadIdx.x;
    if (t >= Etot * 8) return;
    int e = t >> 3, h = t & 7;
    long long d;
    if (e < E) { d = adj[(size_t)E + e]; } else { d = e - E; }
    float v = __expf(eh[t] - segmax[d * 8 + h]);
    eh[t] = v;
    atomicAdd(&denom[d * 8 + h], v);
}

// ---------------------------------------------------------------------------
// Edge pass 3: one wave per edge. conv[dst] += hw[src] * (ex / denom[dst]).
// Lane l covers channel h*32+l for h = 0..7.
// ---------------------------------------------------------------------------
__global__ __launch_bounds__(256) void edge_scatter_kernel(
    const long long* __restrict__ adj, int E, int Etot,
    const float* __restrict__ hw,
    const float* __restrict__ ex,
    const float* __restrict__ denom,
    float* __restrict__ conv)
{
    int gw   = (blockIdx.x * blockDim.x + threadIdx.x) >> 5;
    int lane = threadIdx.x & 31;
    if (gw >= Etot) return;
    long long s, d;
    if (gw < E) { s = adj[gw]; d = adj[(size_t)E + gw]; } else { s = d = gw - E; }
    const float* __restrict__ hrow = hw + (size_t)s * HC;
    float* __restrict__ crow = conv + (size_t)d * HC;
    __builtin_prefetch(hrow + lane, 0, 1);        // global_prefetch_b8
    #pragma unroll
    for (int h = 0; h < 8; ++h) {
        float alpha = ex[(size_t)gw * 8 + h] / denom[d * 8 + h];
        atomicAdd(&crow[h * 32 + lane], hrow[h * 32 + lane] * alpha);
    }
}

// ---------------------------------------------------------------------------
// Layer epilogue: h = ELU(conv + bias + temb); write bf16 A for next GEMM:
// cols 0..255 = h, 256..263 = q, (264..287 remain zero from build_a0).
// ---------------------------------------------------------------------------
__global__ void finalize_layer_kernel(const float* __restrict__ conv,
                                      const float* __restrict__ bias,
                                      const float* __restrict__ temb,
                                      const float* __restrict__ q,
                                      __bf16* __restrict__ Ab, int N)
{
    int t = blockIdx.x * blockDim.x + threadIdx.x;
    if (t >= N * 264) return;
    int n = t / 264;
    int j = t - n * 264;
    float v;
    if (j < HC) v = elu1(conv[(size_t)n * HC + j] + bias[j] + temb[j]);
    else        v = q[(size_t)n * 8 + (j - HC)];
    Ab[(size_t)n * KPAD + j] = (__bf16)v;
}

// ---------------------------------------------------------------------------
// hid = ELU(hid + fin_b1) in place, [N, 528]
// ---------------------------------------------------------------------------
__global__ void elu_bias_kernel(float* __restrict__ hid,
                                const float* __restrict__ b, int N)
{
    int t = blockIdx.x * blockDim.x + threadIdx.x;
    if (t >= N * 528) return;
    int j = t - (t / 528) * 528;
    hid[t] = elu1(hid[t] + b[j]);
}

// ---------------------------------------------------------------------------
// out[n, o] = fin_b2[o] + sum_j hid[n,j] * fin_w2[j,o]   (528 -> 8)
// ---------------------------------------------------------------------------
__global__ void final_proj_kernel(const float* __restrict__ hid,
                                  const float* __restrict__ w2,
                                  const float* __restrict__ b2,
                                  float* __restrict__ out, int N)
{
    int t = blockIdx.x * blockDim.x + threadIdx.x;
    if (t >= N * 8) return;
    int n = t >> 3, o = t & 7;
    const float* __restrict__ hrow = hid + (size_t)n * 528;
    float acc = b2[o];
    #pragma unroll 4
    for (int j = 0; j < 528; ++j) acc += hrow[j] * w2[j * 8 + o];
    out[t] = acc;
}

// ---------------------------------------------------------------------------
// Host side
// ---------------------------------------------------------------------------
extern "C" void kernel_launch(void* const* d_in, const int* in_sizes, int n_in,
                              void* d_out, int out_size, void* d_ws, size_t ws_size,
                              hipStream_t stream)
{
    const float* x      = (const float*)d_in[0];
    const float* q      = (const float*)d_in[1];
    const long long* adj= (const long long*)d_in[2];
    const float* t_in   = (const float*)d_in[3];
    const float* W[3]   = { (const float*)d_in[5], (const float*)d_in[6], (const float*)d_in[7] };
    const int    WK[3]  = { 136, 264, 264 };
    const float* att_s  = (const float*)d_in[8];
    const float* att_d  = (const float*)d_in[9];
    const float* bias   = (const float*)d_in[10];
    const float* tw1    = (const float*)d_in[11];
    const float* tb1    = (const float*)d_in[12];
    const float* tw2    = (const float*)d_in[13];
    const float* tb2    = (const float*)d_in[14];
    const float* fw1    = (const float*)d_in[15];
    const float* fb1    = (const float*)d_in[16];
    const float* fw2    = (const float*)d_in[17];
    const float* fb2    = (const float*)d_in[18];

    const int N    = in_sizes[0] / 128;
    const int E    = in_sizes[2] / 2;
    const int Etot = E + N;

    // ---- workspace bump allocator (256B aligned) ----
    char* ws = (char*)d_ws;
    size_t off = 0;
    auto alloc = [&](size_t bytes) -> char* {
        size_t o = (off + 255) & ~(size_t)255;
        off = o + bytes;
        return ws + o;
    };
    float*  temb   = (float*) alloc(HC * sizeof(float));
    __bf16* Wt[3];
    for (int i = 0; i < 3; ++i) Wt[i] = (__bf16*)alloc((size_t)HC * KPAD * sizeof(__bf16));
    __bf16* FW1t   = (__bf16*)alloc((size_t)528 * KPAD * sizeof(__bf16));
    __bf16* Ab     = (__bf16*)alloc((size_t)N * KPAD * sizeof(__bf16));
    float*  asrc   = (float*) alloc((size_t)N * 8 * sizeof(float));
    float*  adst   = (float*) alloc((size_t)N * 8 * sizeof(float));
    float*  segmax = (float*) alloc((size_t)N * 8 * sizeof(float));
    float*  denom  = (float*) alloc((size_t)N * 8 * sizeof(float));
    float*  eh     = (float*) alloc((size_t)Etot * 8 * sizeof(float));
    float*  hw     = (float*) alloc((size_t)N * HC * sizeof(float));
    float*  conv   = (float*) alloc((size_t)N * HC * sizeof(float));
    (void)  alloc((size_t)N * 16 * sizeof(float));      // tail pad for hid alias
    float*  hid    = hw;  // [N,528] aliases dead hw+conv+pad after GAT layers
    (void)ws_size;

    const int TB = 256;

    // time embedding
    hipLaunchKernelGGL(time_embed_kernel, dim3(1), dim3(128), 0, stream,
                       t_in, tw1, tb1, tw2, tb2, temb);

    // weight conversions (bf16, transposed, K zero-padded to KPAD)
    for (int i = 0; i < 3; ++i) {
        int tot = HC * KPAD;
        hipLaunchKernelGGL(convert_wt_kernel, dim3(cdiv(tot, TB)), dim3(TB), 0, stream,
                           W[i], Wt[i], WK[i], HC, KPAD);
    }
    {
        int tot = 528 * KPAD;
        hipLaunchKernelGGL(convert_wt_kernel, dim3(cdiv(tot, TB)), dim3(TB), 0, stream,
                           fw1, FW1t, 264, 528, KPAD);
    }

    // initial features
    hipLaunchKernelGGL(build_a0_kernel, dim3(cdiv(N * KPAD, TB)), dim3(TB), 0, stream,
                       x, q, Ab, N);

    const dim3 gemm_block(256);
    for (int layer = 0; layer < 3; ++layer) {
        // hw = Ab @ W  (WMMA bf16, async-LDS staged A)
        hipLaunchKernelGGL(wmma_gemm_bf16_kernel,
                           dim3(cdiv(N, 16), cdiv(HC, 128)), gemm_block, 0, stream,
                           Ab, KPAD, Wt[layer], KPAD, hw, HC, N, HC, KPAD);
        // attention logits per node
        hipLaunchKernelGGL(node_alpha_kernel, dim3(cdiv(N * 8, TB)), dim3(TB), 0, stream,
                           hw, att_s + layer * HC, att_d + layer * HC, asrc, adst, N);
        // init accumulators
        hipLaunchKernelGGL(init_accum_kernel, dim3(cdiv(N * HC, TB)), dim3(TB), 0, stream,
                           conv, segmax, denom, N);
        // segment softmax over incoming edges
        hipLaunchKernelGGL(edge_pass1_kernel, dim3(cdiv(Etot * 8, TB)), dim3(TB), 0, stream,
                           adj, E, Etot, asrc, adst, eh, segmax);
        hipLaunchKernelGGL(edge_pass2_kernel, dim3(cdiv(Etot * 8, TB)), dim3(TB), 0, stream,
                           adj, E, Etot, eh, segmax, denom);
        // weighted message scatter (one wave per edge)
        hipLaunchKernelGGL(edge_scatter_kernel, dim3(cdiv(Etot * 32, TB)), dim3(TB), 0, stream,
                           adj, E, Etot, hw, eh, denom, conv);
        // ELU(conv + bias + temb), re-concat q, convert to bf16 for next GEMM
        hipLaunchKernelGGL(finalize_layer_kernel, dim3(cdiv(N * 264, TB)), dim3(TB), 0, stream,
                           conv, bias + layer * HC, temb, q, Ab, N);
    }

    // final MLP: hid = Ab @ fin_w1 (WMMA), ELU(+b1), then 528->8 projection
    hipLaunchKernelGGL(wmma_gemm_bf16_kernel,
                       dim3(cdiv(N, 16), cdiv(528, 128)), gemm_block, 0, stream,
                       Ab, KPAD, FW1t, KPAD, hid, 528, N, 528, KPAD);
    hipLaunchKernelGGL(elu_bias_kernel, dim3(cdiv(N * 528, TB)), dim3(TB), 0, stream,
                       hid, fb1, N);
    hipLaunchKernelGGL(final_proj_kernel, dim3(cdiv(N * 8, TB)), dim3(TB), 0, stream,
                       hid, fw2, fb2, (float*)d_out, N);
}
